// PatchSCD_29240137351693
// MI455X (gfx1250) — matrix operand order repaired and checked by reading, not compiled, not run on hardware
//
#include <hip/hip_runtime.h>
#include <hip/hip_bf16.h>
#include <math.h>

// ---------------- problem constants ----------------
#define PATCHSZ 7
#define NUMPROJ 16
#define NBINS   65536
#define BB      2
#define CCH     3
#define HH      384
#define WWI     384
#define HO      378                 // (384-7)/1+1
#define NPOS    (HO*HO)             // 142884 patch positions
#define NPAD    142896              // NPOS rounded up to multiple of 16
#define KD      147                 // C*7*7
#define KPAD    160                 // K padded; rn zero-padded for k>=147
#define TILES_PER_IMG (NPAD/16)     // 8931
#define TOTAL_TILES   (TILES_PER_IMG*4)  // 2 batches * {x,y}
#define WD      (NBINS-6)           // 65530 valid-conv output width
#define NBX     140                 // blocks per pair for streaming passes

typedef __attribute__((ext_vector_type(2)))  float     v2f;
typedef __attribute__((ext_vector_type(8)))  float     v8f;
typedef __attribute__((ext_vector_type(16))) _Float16  v16h;

#if defined(__has_builtin)
#if __has_builtin(__builtin_amdgcn_wmma_f32_16x16x4_f32)
#define USE_F32_WMMA 1
#endif
#endif

// ---- order-preserving float<->uint map for atomic min/max ----
__device__ __forceinline__ unsigned f2ord(float f) {
    unsigned u = __float_as_uint(f);
    return (u & 0x80000000u) ? ~u : (u | 0x80000000u);
}
__device__ __forceinline__ float ord2f(unsigned s) {
    unsigned u = (s & 0x80000000u) ? (s & 0x7FFFFFFFu) : ~s;
    return __uint_as_float(u);
}

// force a wave-uniform pointer into SGPRs so loads use saddr + 32-bit voffset
__device__ __forceinline__ const float* uniform_cptr(const float* p) {
    unsigned long v = (unsigned long)p;
    unsigned lo = __builtin_amdgcn_readfirstlane((unsigned)v);
    unsigned hi = __builtin_amdgcn_readfirstlane((unsigned)(v >> 32));
    return (const float*)(((unsigned long)hi << 32) | (unsigned long)lo);
}
__device__ __forceinline__ float* uniform_ptr(float* p) {
    unsigned long v = (unsigned long)p;
    unsigned lo = __builtin_amdgcn_readfirstlane((unsigned)v);
    unsigned hi = __builtin_amdgcn_readfirstlane((unsigned)(v >> 32));
    return (float*)(((unsigned long)hi << 32) | (unsigned long)lo);
}

// ---------------- 1) normalize rand by unbiased std over d ----------------
__global__ void prep_rn_kernel(const float* __restrict__ rnd, float* __restrict__ rn) {
    int t = threadIdx.x;
    if (t >= BB * NUMPROJ) return;
    int b = t >> 4, p = t & 15;
    const float* src = rnd + (size_t)b * KD * NUMPROJ + p;
    float mean = 0.0f;
    for (int k = 0; k < KD; ++k) mean += src[(size_t)k * NUMPROJ];
    mean /= (float)KD;
    float var = 0.0f;
    for (int k = 0; k < KD; ++k) { float d = src[(size_t)k * NUMPROJ] - mean; var += d * d; }
    float inv = 1.0f / sqrtf(var / (float)(KD - 1));   // ddof=1
    float* dst = rn + (size_t)b * KPAD * NUMPROJ + p;
    for (int k = 0; k < KD;   ++k) dst[(size_t)k * NUMPROJ] = src[(size_t)k * NUMPROJ] * inv;
    for (int k = KD; k < KPAD; ++k) dst[(size_t)k * NUMPROJ] = 0.0f;  // makes k>=KD A-values inert
}

// ---------------- 2) zero histogram, init min/max cells ----------------
__global__ void zero_init_kernel(float* __restrict__ hist, unsigned* __restrict__ mm) {
    size_t idx = ((size_t)blockIdx.x * blockDim.x + threadIdx.x) * 4;
    if (idx < (size_t)BB * NUMPROJ * NBINS)
        *(float4*)(hist + idx) = make_float4(0.f, 0.f, 0.f, 0.f);
    if (blockIdx.x == 0 && threadIdx.x < 2 * BB * NUMPROJ)
        mm[threadIdx.x] = (threadIdx.x & 1) ? 0u : 0xFFFFFFFFu;   // [min,max] per pair
}

// ---------------- 3) projection GEMM on the WMMA pipe ----------------
// One wave owns a 16(row=n) x 16(col=proj) f32 tile; K-loop over padded 160.
// k->pixel byte-offset table in LDS; B staged in LDS; SGPR image base +
// 32-bit per-lane offsets => 1 VALU add per A load; EXEC all-ones throughout.
__global__ void proj_wmma_kernel(const float* __restrict__ x, const float* __restrict__ y,
                                 const float* __restrict__ rn,
                                 float* __restrict__ projX, float* __restrict__ projY) {
    __shared__ int   koffs[KPAD];                    // k -> byte offset ((c*384+di)*384+dj)*4
    __shared__ float rnL[BB * KPAD * NUMPROJ];       // 2*160*16 floats = 20 KB

    for (int t = threadIdx.x; t < KPAD; t += blockDim.x) {
        int off = 0;
        if (t < KD) {
            int c = t / 49, rem = t - c * 49, di = rem / 7, dj = rem - di * 7;
            off = ((c * HH + di) * WWI + dj) * 4;
        }
        koffs[t] = off;
    }
    for (int t = threadIdx.x; t < BB * KPAD * NUMPROJ; t += blockDim.x)
        rnL[t] = rn[t];
    __syncthreads();

    int wave = threadIdx.x >> 5;
    int lane = threadIdx.x & 31;
    int gw = blockIdx.x * 8 + wave;
    if (gw >= TOTAL_TILES) return;                 // wave-uniform
    int tile = gw % TILES_PER_IMG;
    int bimg = gw / TILES_PER_IMG;
    int b = bimg >> 1, img = bimg & 1;
    const float* src = uniform_cptr((img ? y : x) + (size_t)b * CCH * HH * WWI);
    float*       dst = uniform_ptr((img ? projY : projX) + (size_t)b * NUMPROJ * NPAD);
    const float* rB  = rnL + b * KPAD * NUMPROJ;

    int n0   = tile * 16;
    int half = lane >> 4, l15 = lane & 15;

    // loop-invariant per-lane patch byte base (pad rows clamp to 0: output unused)
    int n = n0 + l15;
    int i = n / HO, j = n - i * HO;
    int pixB = ((n < NPOS) ? (i * WWI + j) : 0) * 4;
    const char* sc = (const char*)src;

    v8f acc = {};
#if defined(USE_F32_WMMA)
    // A 16x4 f32: lanes0-15 -> K = k0,k0+1 ; lanes16-31 -> K = k0+2,k0+3 (M = lane&15)
    // B 4x16 f32: same K striping, N = lane&15
    #pragma unroll 4
    for (int kc = 0; kc < KPAD / 4; ++kc) {
        int k0 = kc * 4 + half * 2;
        v2f a, bf;
        a.x  = *(const float*)(sc + (pixB + koffs[k0]));
        a.y  = *(const float*)(sc + (pixB + koffs[k0 + 1]));
        bf.x = rB[k0 * NUMPROJ + l15];
        bf.y = rB[(k0 + 1) * NUMPROJ + l15];
        acc = __builtin_amdgcn_wmma_f32_16x16x4_f32(false, a, false, bf,
                                                    (short)0, acc, false, false);
    }
#else
    // fallback: confirmed f16 16x16x32 path (f32 accumulate)
    for (int kc = 0; kc < KPAD / 32; ++kc) {
        v16h a, bf;
        int kbA = kc * 32 + half * 8;    // A: e<8 -> K=kbA+e ; e>=8 -> K=kbA+e+8

        #pragma unroll
        for (int e = 0; e < 16; ++e) {
            int ka = kbA + (e < 8 ? e : e + 8);
            a[e]  = (_Float16)(*(const float*)(sc + (pixB + koffs[ka])));
            bf[e] = (_Float16)rB[ka * NUMPROJ + l15];
        }
        acc = __builtin_amdgcn_wmma_f32_16x16x32_f16(false, a, false, bf,
                                                     (short)0, acc, false, false);
    }
#endif
    // C/D layout: lanes0-15 M=vgpr, lanes16-31 M=vgpr+8, N=lane&15.
    // p-major store -> each lane writes 8 consecutive floats (two aligned float4s).
    float* drow = dst + (size_t)l15 * NPAD + n0 + half * 8;
    *(float4*)(drow)     = make_float4(acc[0], acc[1], acc[2], acc[3]);
    *(float4*)(drow + 4) = make_float4(acc[4], acc[5], acc[6], acc[7]);
}

// ---------------- 4) joint min/max over px and py per (b,p) ----------------
__global__ void minmax_kernel(const float* __restrict__ projX, const float* __restrict__ projY,
                              unsigned* __restrict__ mm) {
    int pair = blockIdx.y; int b = pair >> 4, p = pair & 15;
    const float* px = projX + ((size_t)b * NUMPROJ + p) * NPAD;
    const float* py = projY + ((size_t)b * NUMPROJ + p) * NPAD;
    float lmin = INFINITY, lmax = -INFINITY;
    for (int n = blockIdx.x * blockDim.x + threadIdx.x; n < NPOS; n += blockDim.x * gridDim.x) {
        float a = px[n], c = py[n];
        lmin = fminf(lmin, fminf(a, c));
        lmax = fmaxf(lmax, fmaxf(a, c));
    }
    __shared__ float smin[256], smax[256];
    smin[threadIdx.x] = lmin; smax[threadIdx.x] = lmax;
    __syncthreads();
    for (int s = 128; s > 0; s >>= 1) {
        if (threadIdx.x < s) {
            smin[threadIdx.x] = fminf(smin[threadIdx.x], smin[threadIdx.x + s]);
            smax[threadIdx.x] = fmaxf(smax[threadIdx.x], smax[threadIdx.x + s]);
        }
        __syncthreads();
    }
    if (threadIdx.x == 0) {
        atomicMin(&mm[pair * 2 + 0], f2ord(smin[0]));
        atomicMax(&mm[pair * 2 + 1], f2ord(smax[0]));
    }
}

// ---------------- 5) scatter-SET 1.0 into histogram of py ----------------
__global__ void scatter_kernel(const float* __restrict__ projY, const unsigned* __restrict__ mm,
                               float* __restrict__ hist) {
    int pair = blockIdx.y; int b = pair >> 4, p = pair & 15;
    float mn = ord2f(mm[pair * 2]), mx = ord2f(mm[pair * 2 + 1]);
    float scale = mx - mn;
    const float* py = projY + ((size_t)b * NUMPROJ + p) * NPAD;
    float* h = hist + (size_t)pair * NBINS;
    for (int n = blockIdx.x * blockDim.x + threadIdx.x; n < NPOS; n += blockDim.x * gridDim.x) {
        int idx = (int)floorf(((float)(NBINS - 1) * (py[n] - mn)) / scale);
        idx = max(0, min(NBINS - 1, idx));
        h[idx] = 1.0f;                        // set semantics: race-free
    }
}

// ---------------- 6) 7-tap gaussian (valid) + min(.,1) ----------------
__global__ void smooth_kernel(const float* __restrict__ hist, float* __restrict__ smo) {
    int pair = blockIdx.y;
    int w = blockIdx.x * blockDim.x + threadIdx.x;
    if (w >= WD) return;
    const float* h = hist + (size_t)pair * NBINS + w;
    const float sig = 1.2f;
    const float den = -2.0f * sig * sig;
    float s = 0.0f;
    #pragma unroll
    for (int t = 0; t < 7; ++t) {
        float d = (float)(t - 3);
        float g = expf(d * d / den);          // faithful: exp(...) then squared
        g = g * g;
        s += h[t] * g;                        // symmetric kernel: corr == conv
    }
    smo[(size_t)pair * NBINS + w] = fminf(s, 1.0f);
}

// ---------------- 7) bilinear sample at px, per-block partials ----------------
__global__ void sample_kernel(const float* __restrict__ projX, const unsigned* __restrict__ mm,
                              const float* __restrict__ smo, float* __restrict__ part) {
    int pair = blockIdx.y; int b = pair >> 4, p = pair & 15;
    float mn = ord2f(mm[pair * 2]), mx = ord2f(mm[pair * 2 + 1]);
    float scale = mx - mn;
    const float* px = projX + ((size_t)b * NUMPROJ + p) * NPAD;
    const float* sm = smo + (size_t)pair * NBINS;
    float acc = 0.0f;
    for (int n = blockIdx.x * blockDim.x + threadIdx.x; n < NPOS; n += blockDim.x * gridDim.x) {
        float v  = px[n];
        float gx = 2.0f * (v - mn) / scale - 1.0f;
        float ix = ((gx + 1.0f) * (float)WD - 1.0f) * 0.5f;
        float x0 = floorf(ix);
        float w1 = ix - x0;
        int x0i = (int)x0;
        int x1i = x0i + 1;
        float v0 = (x0i >= 0 && x0i < WD) ? sm[x0i] : 0.0f;
        float v1 = (x1i >= 0 && x1i < WD) ? sm[x1i] : 0.0f;
        acc += v0 * (1.0f - w1) + v1 * w1;
    }
    __shared__ float s[256];
    s[threadIdx.x] = acc; __syncthreads();
    for (int k = 128; k > 0; k >>= 1) {
        if (threadIdx.x < k) s[threadIdx.x] += s[threadIdx.x + k];
        __syncthreads();
    }
    if (threadIdx.x == 0) part[(size_t)pair * NBX + blockIdx.x] = s[0];
}

// ---------------- 8) deterministic final reduce: -sum/(B*P) ----------------
__global__ void final_reduce_kernel(const float* __restrict__ part, float* __restrict__ out) {
    float a = 0.0f;
    for (int i = threadIdx.x; i < BB * NUMPROJ * NBX; i += 256) a += part[i];
    __shared__ float s[256];
    s[threadIdx.x] = a; __syncthreads();
    for (int k = 128; k > 0; k >>= 1) {
        if (threadIdx.x < k) s[threadIdx.x] += s[threadIdx.x + k];
        __syncthreads();
    }
    if (threadIdx.x == 0) *out = -s[0] / (float)(BB * NUMPROJ);
}

extern "C" void kernel_launch(void* const* d_in, const int* in_sizes, int n_in,
                              void* d_out, int out_size, void* d_ws, size_t ws_size,
                              hipStream_t stream) {
    const float* x   = (const float*)d_in[0];
    const float* y   = (const float*)d_in[1];
    const float* rnd = (const float*)d_in[2];
    float* out = (float*)d_out;

    // workspace layout (all 16B aligned); total ~53.4 MB, everything L2-resident
    char* ws = (char*)d_ws;
    const size_t RN_OFF   = 0;                                   // 2*160*16*4      = 20480
    const size_t PX_OFF   = RN_OFF + 20480;                      // 2*16*142896*4   = 18290688
    const size_t PY_OFF   = PX_OFF + (size_t)18290688;
    const size_t HIST_OFF = PY_OFF + (size_t)18290688;           // 32*65536*4      = 8388608
    const size_t SMO_OFF  = HIST_OFF + (size_t)8388608;
    const size_t MM_OFF   = SMO_OFF + (size_t)8388608;           // 64 uints
    const size_t PART_OFF = MM_OFF + 256;                        // 32*140 floats

    float*    rn    = (float*)(ws + RN_OFF);
    float*    projX = (float*)(ws + PX_OFF);
    float*    projY = (float*)(ws + PY_OFF);
    float*    hist  = (float*)(ws + HIST_OFF);
    float*    smo   = (float*)(ws + SMO_OFF);
    unsigned* mm    = (unsigned*)(ws + MM_OFF);
    float*    part  = (float*)(ws + PART_OFF);

    prep_rn_kernel<<<1, 32, 0, stream>>>(rnd, rn);
    zero_init_kernel<<<(BB * NUMPROJ * NBINS / 4 + 255) / 256, 256, 0, stream>>>(hist, mm);

    int projBlocks = (TOTAL_TILES + 7) / 8;        // 8 waves (16x16 tiles) per 256-thr block
    proj_wmma_kernel<<<projBlocks, 256, 0, stream>>>(x, y, rn, projX, projY);

    dim3 gPair(NBX, BB * NUMPROJ);
    minmax_kernel<<<gPair, 256, 0, stream>>>(projX, projY, mm);
    scatter_kernel<<<gPair, 256, 0, stream>>>(projY, mm, hist);
    dim3 gSm((WD + 255) / 256, BB * NUMPROJ);
    smooth_kernel<<<gSm, 256, 0, stream>>>(hist, smo);
    sample_kernel<<<gPair, 256, 0, stream>>>(projX, mm, smo, part);
    final_reduce_kernel<<<1, 256, 0, stream>>>(part, out);
}
